// GAE_31250182045963
// MI455X (gfx1250) — matrix operand order repaired and checked by reading, not compiled, not run on hardware
//
#include <hip/hip_runtime.h>
#include <cstdint>

typedef __attribute__((ext_vector_type(16))) __bf16 v16bf;
typedef __attribute__((ext_vector_type(8)))  __bf16 v8bf;
typedef __attribute__((ext_vector_type(4)))  __bf16 v4bf;
typedef __attribute__((ext_vector_type(8)))  float  v8f;
typedef __attribute__((ext_vector_type(4)))  float  v4f;

#define BM 128
#define BN 128
#define BK 64

// ---------------------------------------------------------------------------
// Async global -> LDS copy of one 128x64 bf16 tile (16 KB), 256 threads.
// 16B chunks, XOR-swizzled within each row so later ds_load_b128 fragment
// reads are bank-conflict free. 4 async instructions per thread per tile.
// ---------------------------------------------------------------------------
__device__ __forceinline__ void async_tile_load(const __bf16* gbase, int strideElems,
                                                int rowBase, int kBase,
                                                unsigned ldsBase, int tid) {
#pragma unroll
  for (int i = 0; i < 4; ++i) {
    int chunk = tid + i * 256;          // 0..1023 (128 rows * 8 chunks)
    int row   = chunk >> 3;             // 0..127
    int col   = chunk & 7;              // 16B chunk within a 64-elem row
    unsigned long long ga = (unsigned long long)(uintptr_t)
        (gbase + (long long)(rowBase + row) * strideElems + kBase + col * 8);
    unsigned lds = ldsBase + (unsigned)((row * 8 + (col ^ (row & 7))) * 16);
    asm volatile("global_load_async_to_lds_b128 %0, %1, off"
                 :: "v"(lds), "v"(ga) : "memory");
  }
}

// ---------------------------------------------------------------------------
// Generic:  Out[N][M] = act( P[M][K] * Q[N][K]^T )   (all K-contiguous)
//   P: [gridDim.y*128][K] bf16, row stride sP
//   Q: [gridDim.x*128][K] bf16, row stride sQ
//   Out: [N][M], row stride sOut, bf16 or f32, optional ReLU
// 256 threads = 8 waves; each wave owns a 32(M) x 64(N) patch = 2x4 WMMA tiles.
// ---------------------------------------------------------------------------
template <bool RELU, bool F32OUT>
__global__ void __launch_bounds__(256)
gemm_bf16_wmma(const __bf16* __restrict__ P, const __bf16* __restrict__ Q,
               void* __restrict__ Out, int sP, int sQ, int sOut, int K) {
  __shared__ __bf16 smem[2][2][BM * BK];   // [stage][P/Q][tile] = 64 KB

  const int tid  = threadIdx.x;
  const int lane = tid & 31;
  const int w    = tid >> 5;
  const int wm   = w & 3;        // 4 wave rows  * 32
  const int wn   = w >> 2;       // 2 wave cols  * 64
  const int mBase = blockIdx.y * BM;
  const int nBase = blockIdx.x * BN;
  const int lm = lane & 15;
  const int lh = lane >> 4;

  v8f acc[2][4];
  {
    v8f z;
#pragma unroll
    for (int v = 0; v < 8; ++v) z[v] = 0.0f;
#pragma unroll
    for (int a = 0; a < 2; ++a)
#pragma unroll
      for (int b = 0; b < 4; ++b) acc[a][b] = z;
  }

  unsigned ldsP0 = (unsigned)(uintptr_t)&smem[0][0][0];
  unsigned ldsQ0 = (unsigned)(uintptr_t)&smem[0][1][0];
  unsigned ldsP1 = (unsigned)(uintptr_t)&smem[1][0][0];
  unsigned ldsQ1 = (unsigned)(uintptr_t)&smem[1][1][0];

  const int S = K / BK;
  async_tile_load(P, sP, mBase, 0, ldsP0, tid);
  async_tile_load(Q, sQ, nBase, 0, ldsQ0, tid);

  for (int s = 0; s < S; ++s) {
    const int cur = s & 1;
    if (s + 1 < S) {
      async_tile_load(P, sP, mBase, (s + 1) * BK, cur ? ldsP0 : ldsP1, tid);
      async_tile_load(Q, sQ, nBase, (s + 1) * BK, cur ? ldsQ0 : ldsQ1, tid);
      asm volatile("s_wait_asynccnt 0x8" ::: "memory");  // stage s's 8 done
    } else {
      asm volatile("s_wait_asynccnt 0x0" ::: "memory");
    }
    __syncthreads();   // all waves' stage-s tiles visible

    const __bf16* sP_ = &smem[cur][0][0];
    const __bf16* sQ_ = &smem[cur][1][0];

#pragma unroll
    for (int ks = 0; ks < 2; ++ks) {       // two K=32 WMMA steps per stage
      v16bf afrag[2], bfrag[4];
      // A fragment (16x32 bf16): lane holds row M=lm, K = lh*8+[0,8) and +16
#pragma unroll
      for (int mt = 0; mt < 2; ++mt) {
        int row = wm * 32 + mt * 16 + lm;
        int c0 = ks * 4 + lh;
        int c1 = c0 + 2;
        v8bf lo = *(const v8bf*)&sP_[(row * 8 + (c0 ^ (row & 7))) * 8];
        v8bf hi = *(const v8bf*)&sP_[(row * 8 + (c1 ^ (row & 7))) * 8];
        afrag[mt] = __builtin_shufflevector(lo, hi, 0, 1, 2, 3, 4, 5, 6, 7,
                                            8, 9, 10, 11, 12, 13, 14, 15);
      }
      // B fragment (32x16 bf16): lane holds col N=lm, K = lh*8+[0,8) and +16
#pragma unroll
      for (int nt = 0; nt < 4; ++nt) {
        int row = wn * 64 + nt * 16 + lm;
        int c0 = ks * 4 + lh;
        int c1 = c0 + 2;
        v8bf lo = *(const v8bf*)&sQ_[(row * 8 + (c0 ^ (row & 7))) * 8];
        v8bf hi = *(const v8bf*)&sQ_[(row * 8 + (c1 ^ (row & 7))) * 8];
        bfrag[nt] = __builtin_shufflevector(lo, hi, 0, 1, 2, 3, 4, 5, 6, 7,
                                            8, 9, 10, 11, 12, 13, 14, 15);
      }
#pragma unroll
      for (int mt = 0; mt < 2; ++mt)
#pragma unroll
        for (int nt = 0; nt < 4; ++nt)
          acc[mt][nt] = __builtin_amdgcn_wmma_f32_16x16x32_bf16(
              false, afrag[mt], false, bfrag[nt], (short)0, acc[mt][nt],
              false, false);
    }
    __syncthreads();   // protect LDS stage before it is overwritten
  }

  // Epilogue: D fragment lane holds N=lm, M = lh*8 + v  -> transposed store
  // is 8 contiguous elements: Out[n][m0..m0+7].
#pragma unroll
  for (int mt = 0; mt < 2; ++mt) {
#pragma unroll
    for (int nt = 0; nt < 4; ++nt) {
      v8f a = acc[mt][nt];
      if (RELU) {
#pragma unroll
        for (int v = 0; v < 8; ++v) a[v] = fmaxf(a[v], 0.0f);
      }
      int n  = nBase + wn * 64 + nt * 16 + lm;
      int m0 = mBase + wm * 32 + mt * 16 + lh * 8;
      if (F32OUT) {
        float* o = (float*)Out + (long long)n * sOut + m0;
        v4f lo4 = {a[0], a[1], a[2], a[3]};
        v4f hi4 = {a[4], a[5], a[6], a[7]};
        *(v4f*)o = lo4;
        *(v4f*)(o + 4) = hi4;
      } else {
        __bf16* o = (__bf16*)Out + (long long)n * sOut + m0;
        v8bf r;
#pragma unroll
        for (int v = 0; v < 8; ++v) r[v] = (__bf16)a[v];
        *(v8bf*)o = r;
      }
    }
  }
}

// ---------------------------------------------------------------------------
// Prep kernels
// ---------------------------------------------------------------------------
__global__ void cvt_f32_bf16_v4(const float* __restrict__ s,
                                __bf16* __restrict__ d, long long n4) {
  long long i = blockIdx.x * (long long)blockDim.x + threadIdx.x;
  long long stride = (long long)gridDim.x * blockDim.x;
  for (; i < n4; i += stride) {
    v4f x = ((const v4f*)s)[i];
    v4bf y;
#pragma unroll
    for (int v = 0; v < 4; ++v) y[v] = (__bf16)x[v];
    ((v4bf*)d)[i] = y;
  }
}

__global__ void transpose_w_bf16(const float* __restrict__ W,
                                 __bf16* __restrict__ Wt, int Kd, int Jd) {
  int idx = blockIdx.x * blockDim.x + threadIdx.x;
  if (idx < Kd * Jd) {
    int k = idx / Jd, j = idx % Jd;
    Wt[(long long)j * Kd + k] = (__bf16)W[idx];
  }
}

// ---------------------------------------------------------------------------
// Host launcher
// ---------------------------------------------------------------------------
extern "C" void kernel_launch(void* const* d_in, const int* in_sizes, int n_in,
                              void* d_out, int out_size, void* d_ws, size_t ws_size,
                              hipStream_t stream) {
  (void)in_sizes; (void)n_in; (void)out_size; (void)ws_size;
  const int Nn = 8192;
  const float* X   = (const float*)d_in[0];
  const float* adj = (const float*)d_in[1];
  const float* Wf[6] = {(const float*)d_in[2], (const float*)d_in[3],
                        (const float*)d_in[4], (const float*)d_in[5],
                        (const float*)d_in[6], (const float*)d_in[7]};
  const int dins[6]  = {512, 256, 256, 128, 256, 256};
  const int douts[6] = {256, 256, 128, 256, 256, 512};

  // Workspace layout (bf16): adj (134 MB, fits L2!) | Z | T | Wt[6]
  char* ws = (char*)d_ws;
  __bf16* adjb = (__bf16*)ws;                       size_t off = (size_t)Nn * Nn * 2;
  __bf16* Zbuf = (__bf16*)(ws + off);               off += (size_t)Nn * 512 * 2;
  __bf16* Tbuf = (__bf16*)(ws + off);               off += (size_t)Nn * 512 * 2;
  __bf16* Wt[6];
  for (int i = 0; i < 6; ++i) {
    Wt[i] = (__bf16*)(ws + off);
    off += (size_t)dins[i] * douts[i] * 2;
  }

  // Prep: adj -> bf16, X -> bf16, W -> bf16 transposed (Wt[j][k])
  cvt_f32_bf16_v4<<<4096, 256, 0, stream>>>(adj, adjb, (long long)Nn * Nn / 4);
  cvt_f32_bf16_v4<<<1024, 256, 0, stream>>>(X, Zbuf, (long long)Nn * 512 / 4);
  for (int i = 0; i < 6; ++i) {
    int n = dins[i] * douts[i];
    transpose_w_bf16<<<(n + 255) / 256, 256, 0, stream>>>(Wf[i], Wt[i], dins[i], douts[i]);
  }

  for (int i = 0; i < 6; ++i) {
    const int di = dins[i], dr = douts[i];
    // Kernel A: Tt[j][node] = (Z * Wt^T) transposed-store.
    //   M = 8192 nodes (P=Z, stride di), N = dr (Q=Wt, stride di), K = di.
    gemm_bf16_wmma<false, false>
        <<<dim3(dr / BN, Nn / BM), 256, 0, stream>>>(
            Zbuf, Wt[i], (void*)Tbuf, di, di, Nn, di);
    // Kernel B: Z'[node][f] = relu(Tt * adj^T) transposed-store.
    //   M = dr features (P=Tt, stride 8192), N = 8192 nodes (Q=adjb), K = 8192.
    if (i < 5) {
      gemm_bf16_wmma<true, false>
          <<<dim3(Nn / BN, dr / BM), 256, 0, stream>>>(
              Tbuf, adjb, (void*)Zbuf, Nn, Nn, dr, Nn);
    } else {
      gemm_bf16_wmma<true, true>
          <<<dim3(Nn / BN, dr / BM), 256, 0, stream>>>(
              Tbuf, adjb, d_out, Nn, Nn, dr, Nn);
    }
  }
}
